// SelfAttention_3238405341596
// MI455X (gfx1250) — matrix operand order repaired and checked by reading, not compiled, not run on hardware
//
#include <hip/hip_runtime.h>

// Self-attention (B=1, S=4096, D=512, H=8, HD=64) for gfx1250.
// Pipeline:
//   1) mask_pack : int32 mask [S,S] -> bitmask [S, S/32]  (64MB read once -> 2MB, L2-resident)
//   2) proj      : per-head Linear for q,k,v (f32 VALU, tiny FLOPs) -> f16; V stored transposed
//   3) attn      : flash-attention, v_wmma_f32_16x16x32_f16 for QK^T and P*V, f32 online softmax
//   4) cvt+fc    : 4096x512 @ 512x512^T via WMMA f16, f32 out + bias
#define Sdim 4096
#define Dm   512
#define Hn   8
#define HD   64
#define SW   (Sdim/32)   // mask words per row

typedef __attribute__((ext_vector_type(16))) _Float16 v16h;
typedef __attribute__((ext_vector_type(8)))  _Float16 v8h;
typedef __attribute__((ext_vector_type(8)))  float    v8f;

static __device__ __forceinline__ v16h cat8(v8h a, v8h b) {
  return __builtin_shufflevector(a, b, 0,1,2,3,4,5,6,7,8,9,10,11,12,13,14,15);
}
static __device__ __forceinline__ v8f wmma_f16(v16h a, v16h b, v8f c) {
  // D = A(16x32 f16) * B(32x16 f16) + C(16x16 f32)
  return __builtin_amdgcn_wmma_f32_16x16x32_f16(false, a, false, b, (short)0, c, false, false);
}

// ---------------------------------------------------------------------------
// 1) Pack int32 mask into bits via wave32 ballot. Each wave -> 1024 elems / 32 words.
__global__ void mask_pack_kernel(const int* __restrict__ mask, unsigned* __restrict__ mp) {
  const int lane = threadIdx.x & 31;
  const long waveId = (long)((blockIdx.x * blockDim.x + threadIdx.x) >> 5);
  const long base = waveId * 1024;
  unsigned myword = 0;
  for (int i = 0; i < 32; ++i) {
    int m = mask[base + i * 32 + lane];
    unsigned long long b = __ballot(m != 0);
    if (lane == i) myword = (unsigned)b;
  }
  mp[base / 32 + lane] = myword;
}

// ---------------------------------------------------------------------------
// 2) Per-head projections. Block = 256 thr handles (head h, 64 rows).
//    qh,kh: [H][S][64] f16 row-major.  vT: [H][64][S] f16 (transposed for B-fragments).
__global__ void proj_kernel(const float* __restrict__ qin, const float* __restrict__ kin,
                            const float* __restrict__ vin,
                            const float* __restrict__ Wq, const float* __restrict__ bq,
                            const float* __restrict__ Wk, const float* __restrict__ bk,
                            const float* __restrict__ Wv, const float* __restrict__ bv,
                            _Float16* __restrict__ qh, _Float16* __restrict__ kh,
                            _Float16* __restrict__ vT) {
  __shared__ _Float16 lv[64][66];
  const int t  = threadIdx.x;
  const int h  = blockIdx.x / (Sdim / 64);
  const int s0 = (blockIdx.x % (Sdim / 64)) * 64;
  const int d  = t & 63;
  const int rg = t >> 6;  // 0..3
  const float* wqr = Wq + d * HD;
  const float* wkr = Wk + d * HD;
  const float* wvr = Wv + d * HD;
  for (int i = 0; i < 16; ++i) {
    const int rl = rg * 16 + i;
    const int s  = s0 + rl;
    const float* xq = qin + (size_t)s * Dm + h * HD;
    const float* xk = kin + (size_t)s * Dm + h * HD;
    const float* xv = vin + (size_t)s * Dm + h * HD;
    float aq = bq[d], ak = bk[d], av = bv[d];
#pragma unroll 8
    for (int j = 0; j < HD; ++j) {
      aq += xq[j] * wqr[j];
      ak += xk[j] * wkr[j];
      av += xv[j] * wvr[j];
    }
    qh[((size_t)h * Sdim + s) * HD + d] = (_Float16)aq;
    kh[((size_t)h * Sdim + s) * HD + d] = (_Float16)ak;
    lv[rl][d] = (_Float16)av;
  }
  __syncthreads();
  const int dd = t >> 2;         // 0..63
  const int sc = (t & 3) * 16;   // 0,16,32,48
  _Float16* dst = vT + ((size_t)h * HD + dd) * Sdim + s0 + sc;
#pragma unroll
  for (int i = 0; i < 16; ++i) dst[i] = lv[sc + i][dd];
}

// ---------------------------------------------------------------------------
// 3) Flash attention. Block = 8 waves; wave w == head w, 32 query rows / block.
//    Per 32-key block: 8 WMMA for S=QK^T, mask+scale, online softmax, 8 WMMA for P*V.
__global__ void __launch_bounds__(256) attn_kernel(const _Float16* __restrict__ qh,
                                                   const _Float16* __restrict__ kh,
                                                   const _Float16* __restrict__ vT,
                                                   const unsigned* __restrict__ mp,
                                                   _Float16* __restrict__ attnOut) {
  __shared__ _Float16 lds_p[8][32][40];  // per-wave P tile, rows 16B aligned (stride 80B)
  const int lane = threadIdx.x & 31;
  const int w    = threadIdx.x >> 5;   // wave id == head
  const int h    = w;
  const int half = lane >> 4;
  const int lh   = lane & 15;
  const int q0   = blockIdx.x * 32;

  // Q A-fragments: [mtile][ksplit], A 16x32: lanes 0-15 K={0..15}, lanes 16-31 K={16..31}
  v16h qa[2][2];
#pragma unroll
  for (int m = 0; m < 2; ++m) {
    const _Float16* qp = qh + ((size_t)h * Sdim + q0 + m * 16 + lh) * HD;
#pragma unroll
    for (int ks = 0; ks < 2; ++ks) {
      v8h lo = *(const v8h*)(qp + ks * 32 + 16 * half);
      v8h hi = *(const v8h*)(qp + ks * 32 + 16 * half + 8);
      qa[m][ks] = cat8(lo, hi);
    }
  }

  v8f o[2][4];
#pragma unroll
  for (int m = 0; m < 2; ++m)
#pragma unroll
    for (int n = 0; n < 4; ++n) o[m][n] = {};
  float mrow[2][8], lrow[2][8];
#pragma unroll
  for (int m = 0; m < 2; ++m)
#pragma unroll
    for (int r = 0; r < 8; ++r) { mrow[m][r] = -3.0e38f; lrow[m][r] = 0.0f; }

  const unsigned* mprow = mp + (size_t)(q0 + lane) * SW;  // lane l caches mask row q0+l

  for (int kb = 0; kb < Sdim / 32; ++kb) {
    const int c0 = kb * 32;
    // ---- S = Q * K^T for 32 key columns (2 n-tiles x 2 k-splits) ----
    v8f sfr[2][2];
#pragma unroll
    for (int nt = 0; nt < 2; ++nt) {
      const _Float16* kp = kh + ((size_t)h * Sdim + c0 + nt * 16 + lh) * HD + 16 * half;
      v16h b0 = *(const v16h*)(kp);
      v16h b1 = *(const v16h*)(kp + 32);
#pragma unroll
      for (int m = 0; m < 2; ++m) {
        v8f z = {};
        z = wmma_f16(qa[m][0], b0, z);
        z = wmma_f16(qa[m][1], b1, z);
        sfr[m][nt] = z;
      }
    }
    const unsigned wq = mprow[kb];
    // ---- mask (before scale), online softmax ----
#pragma unroll
    for (int m = 0; m < 2; ++m) {
      float t0[8], t1[8];
#pragma unroll
      for (int r = 0; r < 8; ++r) {
        const unsigned wm = (unsigned)__shfl((int)wq, m * 16 + r + 8 * half);
        const float v0 = sfr[m][0][r];
        const float v1 = sfr[m][1][r];
        t0[r] = ((wm >> lh) & 1u)        ? v0 * 0.125f : -1.25e19f;  // (-1e20)/8
        t1[r] = ((wm >> (16 + lh)) & 1u) ? v1 * 0.125f : -1.25e19f;
      }
      float al[8];
#pragma unroll
      for (int r = 0; r < 8; ++r) {
        float tm = fmaxf(t0[r], t1[r]);
#pragma unroll
        for (int off = 8; off >= 1; off >>= 1) tm = fmaxf(tm, __shfl_xor(tm, off));
        const float mn = fmaxf(mrow[m][r], tm);
        al[r] = __expf(mrow[m][r] - mn);
        mrow[m][r] = mn;
        const float p0 = __expf(t0[r] - mn);
        const float p1 = __expf(t1[r] - mn);
        float rs = p0 + p1;
#pragma unroll
        for (int off = 8; off >= 1; off >>= 1) rs += __shfl_xor(rs, off);
        lrow[m][r] = lrow[m][r] * al[r] + rs;
        lds_p[w][m * 16 + r + 8 * half][lh]      = (_Float16)p0;
        lds_p[w][m * 16 + r + 8 * half][16 + lh] = (_Float16)p1;
      }
#pragma unroll
      for (int n = 0; n < 4; ++n)
#pragma unroll
        for (int r = 0; r < 8; ++r) o[m][n][r] *= al[r];
    }
    // ---- P A-fragments from LDS (C-layout -> A-layout transpose) ----
    v16h pa[2];
#pragma unroll
    for (int m = 0; m < 2; ++m) {
      const _Float16* pp = &lds_p[w][m * 16 + lh][0];
      v8h lo = *(const v8h*)(pp + 16 * half);
      v8h hi = *(const v8h*)(pp + 16 * half + 8);
      pa[m] = cat8(lo, hi);
    }
    // ---- O += P * V  (V B-fragments contiguous thanks to vT layout) ----
#pragma unroll
    for (int n = 0; n < 4; ++n) {
      const _Float16* vp = vT + ((size_t)h * HD + n * 16 + lh) * Sdim + c0 + 16 * half;
      v16h vb = *(const v16h*)vp;
#pragma unroll
      for (int m = 0; m < 2; ++m) o[m][n] = wmma_f16(pa[m], vb, o[m][n]);
    }
    // ---- prefetch next K/V tile (global_prefetch_b8) ----
    if (kb + 1 < Sdim / 32) {
      __builtin_prefetch(kh + ((size_t)h * Sdim + c0 + 32 + lane) * HD, 0, 1);
      __builtin_prefetch(vT + ((size_t)h * HD + lane * 2) * Sdim + c0 + 32, 0, 1);
    }
  }
  // ---- normalize and store to attnOut[s][h*64+d] (f16 feeds the FC WMMA) ----
#pragma unroll
  for (int m = 0; m < 2; ++m) {
#pragma unroll
    for (int r = 0; r < 8; ++r) {
      const float rl = 1.0f / lrow[m][r];
      const int row = q0 + m * 16 + r + 8 * half;
      _Float16* op = attnOut + (size_t)row * Dm + h * HD + lh;
#pragma unroll
      for (int n = 0; n < 4; ++n) op[n * 16] = (_Float16)(o[m][n][r] * rl);
    }
  }
}

// ---------------------------------------------------------------------------
// 4a) f32 -> f16 weight conversion for the FC
__global__ void cvt_kernel(const float* __restrict__ src, _Float16* __restrict__ dst, int n) {
  int i = blockIdx.x * blockDim.x + threadIdx.x;
  if (i < n) dst[i] = (_Float16)src[i];
}

// 4b) out = x @ Wfc^T + bfc via WMMA. Wave handles a 16x64 output stripe.
__global__ void __launch_bounds__(256) fc_kernel(const _Float16* __restrict__ x16,
                                                 const _Float16* __restrict__ wfc,
                                                 const float* __restrict__ bfc,
                                                 float* __restrict__ out) {
  const int lane = threadIdx.x & 31;
  const int w    = threadIdx.x >> 5;
  const int gid  = blockIdx.x * 8 + w;
  const int rowTile = gid >> 3;
  const int cs      = gid & 7;
  const int half = lane >> 4, lh = lane & 15;
  const int r0 = rowTile * 16, c0 = cs * 64;
  v8f o[4];
#pragma unroll
  for (int n = 0; n < 4; ++n) o[n] = {};
  for (int k0 = 0; k0 < Dm; k0 += 32) {
    const _Float16* xp = x16 + (size_t)(r0 + lh) * Dm + k0;
    v16h a = cat8(*(const v8h*)(xp + 16 * half), *(const v8h*)(xp + 16 * half + 8));
#pragma unroll
    for (int n = 0; n < 4; ++n) {
      const _Float16* wp = wfc + (size_t)(c0 + n * 16 + lh) * Dm + k0 + 16 * half;
      v16h b = *(const v16h*)wp;
      o[n] = wmma_f16(a, b, o[n]);
    }
  }
#pragma unroll
  for (int n = 0; n < 4; ++n) {
    const int col = c0 + n * 16 + lh;
    const float bb = bfc[col];
#pragma unroll
    for (int r = 0; r < 8; ++r) {
      const int row = r0 + r + 8 * half;
      out[(size_t)row * Dm + col] = o[n][r] + bb;
    }
  }
}

// ---------------------------------------------------------------------------
extern "C" void kernel_launch(void* const* d_in, const int* in_sizes, int n_in,
                              void* d_out, int out_size, void* d_ws, size_t ws_size,
                              hipStream_t stream) {
  (void)in_sizes; (void)n_in; (void)out_size; (void)ws_size;
  const float* query = (const float*)d_in[0];
  const float* key_  = (const float*)d_in[1];
  const float* value = (const float*)d_in[2];
  const int*   mask  = (const int*)d_in[3];
  const float* Wq = (const float*)d_in[4];
  const float* bq = (const float*)d_in[5];
  const float* Wk = (const float*)d_in[6];
  const float* bk = (const float*)d_in[7];
  const float* Wv = (const float*)d_in[8];
  const float* bv = (const float*)d_in[9];
  const float* Wfc = (const float*)d_in[10];
  const float* bfc = (const float*)d_in[11];
  float* out = (float*)d_out;

  char* ws = (char*)d_ws;
  _Float16* qh    = (_Float16*)(ws);                   // 4 MB  [H][S][64]
  _Float16* kh    = (_Float16*)(ws + (4u  << 20));     // 4 MB  [H][S][64]
  _Float16* vT    = (_Float16*)(ws + (8u  << 20));     // 4 MB  [H][64][S]
  _Float16* x16   = (_Float16*)(ws + (12u << 20));     // 4 MB  [S][512]
  unsigned* mp    = (unsigned*)(ws + (16u << 20));     // 2 MB  [S][S/32]
  _Float16* wfc16 = (_Float16*)(ws + (18u << 20));     // 0.5 MB

  mask_pack_kernel<<<(Sdim * Sdim / 1024) / 8, 256, 0, stream>>>(mask, mp);
  proj_kernel<<<Hn * (Sdim / 64), 256, 0, stream>>>(query, key_, value,
                                                    Wq, bq, Wk, bk, Wv, bv, qh, kh, vT);
  cvt_kernel<<<(Dm * Dm + 255) / 256, 256, 0, stream>>>(Wfc, wfc16, Dm * Dm);
  attn_kernel<<<Sdim / 32, 256, 0, stream>>>(qh, kh, vT, mp, x16);
  fc_kernel<<<(Sdim / 16) * (Dm / 64) / 8, 256, 0, stream>>>(x16, wfc16, bfc, out);
}
